// VectorQuantizerSingle_33535104647394
// MI455X (gfx1250) — matrix operand order repaired and checked by reading, not compiled, not run on hardware
//
#include <hip/hip_runtime.h>
#include <hip/hip_bf16.h>

typedef _Float16 v8h  __attribute__((ext_vector_type(8)));
typedef _Float16 v16h __attribute__((ext_vector_type(16)));
typedef float    v8f  __attribute__((ext_vector_type(8)));
typedef unsigned int u32x4 __attribute__((ext_vector_type(4)));
typedef int          i32x4 __attribute__((ext_vector_type(4)));
typedef int          i32x8 __attribute__((ext_vector_type(8)));

#define B_DIM 32
#define D_DIM 256
#define T_DIM 1024
#define K_CODES 8192
#define BT (B_DIM * T_DIM)           // 32768
#define BDT (B_DIM * D_DIM * T_DIM)  // 8388608
#define EMB_SCALE 8192.0f
#define LDS_STRIDE 264  // halfs -> 528B rows: optimal (2-way) bank spread for ds_load_b128
#define BLOCK_M 128     // rows of x per block (4 M-waves x 32 rows)
#define STEP_N 64       // codebook rows staged in LDS per step (2 N-waves x 32 cols)
#define N_STEPS (K_CODES / STEP_N)  // 128

#if defined(__gfx1250__) && __has_builtin(__builtin_amdgcn_tensor_load_to_lds) && \
    __has_builtin(__builtin_amdgcn_s_wait_tensorcnt)
#define USE_TDM 1
#else
#define USE_TDM 0
#endif

// ---------------------------------------------------------------------------
// TDM: 2D tile of f16 [tile_rows x 256] -> LDS, row-padded by 16B so the LDS
// row stride is 528B (= LDS_STRIDE halfs). D# per CDNA5 ISA.
// This toolchain exposes the 6-arg builtin:
//   (u32x4 g0, i32x8 g1, i32x4 g2, i32x4 g3, i32x8 extra, i32 cpol)
// ---------------------------------------------------------------------------
#if USE_TDM
__device__ __forceinline__ void tdm_load_tile_f16(unsigned lds_byte_off,
                                                  const _Float16* gsrc,
                                                  unsigned tile_rows,
                                                  unsigned tensor_rows) {
  const unsigned long long ga = (unsigned long long)(uintptr_t)gsrc;
  u32x4 g0;
  g0[0] = 1u;                   // count=1, user descriptor, no gather
  g0[1] = lds_byte_off;         // lds_addr
  g0[2] = (unsigned)ga;         // global_addr[31:0]
  g0[3] = (unsigned)((ga >> 32) & 0x01FFFFFFu) | (2u << 30);  // addr[56:32] | type=2
  i32x8 g1;
  // data_size=2B(1), pad_enable, pad_interval=6 (128 DWORDs=512B row), pad_amount=3 (4 DWORDs=16B)
  g1[0] = (1 << 16) | (1 << 20) | (6 << 22) | (3 << 25);
  g1[1] = (int)((unsigned)D_DIM << 16);                  // tensor_dim0[15:0] = 256
  g1[2] = (int)((tensor_rows & 0xFFFFu) << 16);          // tensor_dim0 hi=0 | tensor_dim1 lo
  g1[3] = (int)((tensor_rows >> 16) & 0xFFFFu) | (int)((unsigned)D_DIM << 16);  // dim1 hi | tile_dim0=256
  g1[4] = (int)(tile_rows & 0xFFFFu);                    // tile_dim1 | tile_dim2=0
  g1[5] = D_DIM;                                         // tensor_dim0_stride = 256 elems
  g1[6] = 0;                                             // stride hi | tensor_dim1_stride lo
  g1[7] = 0;
  const i32x4 z4 = {0, 0, 0, 0};
  const i32x8 z8 = {0, 0, 0, 0, 0, 0, 0, 0};
  __builtin_amdgcn_tensor_load_to_lds(g0, g1, z4, z4, z8, 0);
}
#endif

// fragment loaders matching ISA VGPR layouts (CDNA5 ISA §7.12.2)
__device__ __forceinline__ v16h ld_frag_a(const _Float16* p) {  // chunks at +0 and +16
  const v8h lo = *(const v8h*)p;
  const v8h hi = *(const v8h*)(p + 16);
  return __builtin_shufflevector(lo, hi, 0, 1, 2, 3, 4, 5, 6, 7, 8, 9, 10, 11, 12, 13, 14, 15);
}
__device__ __forceinline__ v16h ld_frag_b(const _Float16* p) {  // chunks at +0 and +16
  const v8h lo = *(const v8h*)p;
  const v8h hi = *(const v8h*)(p + 16);
  return __builtin_shufflevector(lo, hi, 0, 1, 2, 3, 4, 5, 6, 7, 8, 9, 10, 11, 12, 13, 14, 15);
}

// ---------------------------------------------------------------------------
// Kernel 1a: z_e [B,D,T] f32 -> xT [B*T, D] f16 (LDS-tiled transpose)
// ---------------------------------------------------------------------------
__global__ __launch_bounds__(256) void vq_xpose_kernel(const float* __restrict__ z_e,
                                                       _Float16* __restrict__ xT) {
  __shared__ float tile[32][33];
  const int b  = blockIdx.z;
  const int d0 = blockIdx.y * 32;
  const int t0 = blockIdx.x * 32;
  const int tx = threadIdx.x;
  const int ty = threadIdx.y;
#pragma unroll
  for (int k = 0; k < 32; k += 8)
    tile[ty + k][tx] = z_e[b * (D_DIM * T_DIM) + (d0 + ty + k) * T_DIM + t0 + tx];
  __syncthreads();
#pragma unroll
  for (int k = 0; k < 32; k += 8)
    xT[((size_t)b * T_DIM + t0 + ty + k) * D_DIM + d0 + tx] = (_Float16)tile[tx][ty + k];
}

// ---------------------------------------------------------------------------
// Kernel 1b: emb -> emb_f16 * 8192 (keep out of f16 denormal range);
//            c_k = 0.5 * S * ||e_k||^2  so argmax(P - c) == argmin L2 dist
// ---------------------------------------------------------------------------
__global__ __launch_bounds__(256) void vq_embprep_kernel(const float* __restrict__ emb,
                                                         _Float16* __restrict__ ef16,
                                                         float* __restrict__ cbuf) {
  __shared__ float sred[256];
  const int k = blockIdx.x;
  const int tid = threadIdx.x;
  const float v = emb[k * D_DIM + tid];
  ef16[k * D_DIM + tid] = (_Float16)(v * EMB_SCALE);
  sred[tid] = v * v;
  __syncthreads();
  for (int s = 128; s > 0; s >>= 1) {
    if (tid < s) sred[tid] += sred[tid + s];
    __syncthreads();
  }
  if (tid == 0) cbuf[k] = 0.5f * EMB_SCALE * sred[0];
}

// ---------------------------------------------------------------------------
// Kernel 2: WMMA GEMM + fused argmax.
//   block = 128 M-rows, 8 waves as 4(M) x 2(N); wave tile 32M x 32N per step.
//   x tile (128x256 f16) TDM-loaded once; emb tiles (64x256 f16) TDM
//   double-buffered: wave 0 issues tensor_load_to_lds + s_wait_tensorcnt,
//   workgroup barrier hands the buffer to all waves.
// ---------------------------------------------------------------------------
__global__ __launch_bounds__(256) void vq_argmax_kernel(const _Float16* __restrict__ xT,
                                                        const _Float16* __restrict__ ef16,
                                                        const float* __restrict__ cbuf,
                                                        int* __restrict__ idxbuf,
                                                        float* __restrict__ idx_out) {
  __shared__ __align__(16) _Float16 xs[BLOCK_M * LDS_STRIDE];
  __shared__ __align__(16) _Float16 bs[2][STEP_N * LDS_STRIDE];
  __shared__ float redv[BLOCK_M][2];
  __shared__ int   redi[BLOCK_M][2];

  const int tid = threadIdx.x;
  const int gmBase = blockIdx.x * BLOCK_M;
  const int wid  = tid >> 5;
  const int lane = tid & 31;
  const int wm   = wid >> 1;   // 0..3: M-wave (32 rows each)
  const int wn   = wid & 1;    // 0..1: N-wave (32 cols each)
  const int half = lane >> 4;
  const int l16  = lane & 15;

#if USE_TDM
  if (wid == 0) {
    tdm_load_tile_f16((unsigned)(uintptr_t)&xs[0], xT + (size_t)gmBase * D_DIM, BLOCK_M, BT);
    tdm_load_tile_f16((unsigned)(uintptr_t)&bs[0][0], ef16, STEP_N, K_CODES);
  }
#else
  for (int c = tid; c < BLOCK_M * (D_DIM / 8); c += 256) {
    const int row = c >> 5;
    const int col = (c & 31) << 3;
    *(v8h*)&xs[row * LDS_STRIDE + col] = *(const v8h*)&xT[((size_t)gmBase + row) * D_DIM + col];
  }
#endif

  float bestv[4][8];
  int   besti[4][8];
#pragma unroll
  for (int t = 0; t < 4; ++t)
#pragma unroll
    for (int r = 0; r < 8; ++r) { bestv[t][r] = -3.4e38f; besti[t][r] = 0; }

  // A fragment bases (row = wm*32 + mt*16 + l16; K chunks at half*8 and half*8+16)
  const _Float16* aB0 = &xs[(wm * 32 + l16) * LDS_STRIDE + half * 8];
  const _Float16* aB1 = aB0 + 16 * LDS_STRIDE;

  for (int s = 0; s < N_STEPS; ++s) {
#if USE_TDM
    if (wid == 0) __builtin_amdgcn_s_wait_tensorcnt(0);
    __syncthreads();  // buffer s&1 ready for everyone; prev step's reads done
    if (wid == 0 && s + 1 < N_STEPS)
      tdm_load_tile_f16((unsigned)(uintptr_t)&bs[(s + 1) & 1][0],
                        ef16 + (size_t)(s + 1) * STEP_N * D_DIM, STEP_N, K_CODES);
    const _Float16* bt = &bs[s & 1][0];
#else
    __syncthreads();
    for (int c = tid; c < STEP_N * (D_DIM / 8); c += 256) {
      const int row = c >> 5;
      const int col = (c & 31) << 3;
      *(v8h*)&bs[0][row * LDS_STRIDE + col] =
          *(const v8h*)&ef16[((size_t)s * STEP_N + row) * D_DIM + col];
    }
    __syncthreads();
    const _Float16* bt = &bs[0][0];
#endif

    // B fragment bases (col n = wn*32 + nt*16 + l16; K chunks at half*8 and half*8+16)
    const _Float16* bB0 = &bt[(wn * 32 + l16) * LDS_STRIDE + half * 8];
    const _Float16* bB1 = bB0 + 16 * LDS_STRIDE;

    v8f acc00 = {}, acc01 = {}, acc10 = {}, acc11 = {};
#pragma unroll
    for (int dk = 0; dk < D_DIM; dk += 32) {
      const v16h af0 = ld_frag_a(aB0 + dk);
      const v16h af1 = ld_frag_a(aB1 + dk);
      const v16h bf0 = ld_frag_b(bB0 + dk);
      const v16h bf1 = ld_frag_b(bB1 + dk);
      acc00 = __builtin_amdgcn_wmma_f32_16x16x32_f16(false, af0, false, bf0, (short)0, acc00,
                                                     false, false);
      acc01 = __builtin_amdgcn_wmma_f32_16x16x32_f16(false, af0, false, bf1, (short)0, acc01,
                                                     false, false);
      acc10 = __builtin_amdgcn_wmma_f32_16x16x32_f16(false, af1, false, bf0, (short)0, acc10,
                                                     false, false);
      acc11 = __builtin_amdgcn_wmma_f32_16x16x32_f16(false, af1, false, bf1, (short)0, acc11,
                                                     false, false);
    }

    const int n0 = s * STEP_N + wn * 32 + l16;
    const int n1 = n0 + 16;
    const float cv0 = cbuf[n0];
    const float cv1 = cbuf[n1];
#pragma unroll
    for (int r = 0; r < 8; ++r) {
      float sc;
      sc = acc00[r] - cv0; if (sc > bestv[0][r]) { bestv[0][r] = sc; besti[0][r] = n0; }
      sc = acc01[r] - cv1; if (sc > bestv[1][r]) { bestv[1][r] = sc; besti[1][r] = n1; }
      sc = acc10[r] - cv0; if (sc > bestv[2][r]) { bestv[2][r] = sc; besti[2][r] = n0; }
      sc = acc11[r] - cv1; if (sc > bestv[3][r]) { bestv[3][r] = sc; besti[3][r] = n1; }
    }
  }

  // merge nt pair, reduce across the 16 N-lanes (C layout: slot r <-> row +half*8+r)
#pragma unroll
  for (int mt = 0; mt < 2; ++mt) {
#pragma unroll
    for (int r = 0; r < 8; ++r) {
      float v = bestv[mt * 2 + 0][r];
      int   i = besti[mt * 2 + 0][r];
      const float v1 = bestv[mt * 2 + 1][r];
      const int   i1 = besti[mt * 2 + 1][r];
      if (v1 > v || (v1 == v && i1 < i)) { v = v1; i = i1; }
#pragma unroll
      for (int off = 8; off >= 1; off >>= 1) {
        const float ov = __shfl_xor(v, off, 32);
        const int   oi = __shfl_xor(i, off, 32);
        if (ov > v || (ov == v && oi < i)) { v = ov; i = oi; }
      }
      if (l16 == 0) {
        const int row = wm * 32 + mt * 16 + half * 8 + r;
        redv[row][wn] = v;
        redi[row][wn] = i;
      }
    }
  }
  __syncthreads();

  if (tid < BLOCK_M) {
    float v = redv[tid][0];
    int   i = redi[tid][0];
    const float ov = redv[tid][1];
    const int   oi = redi[tid][1];
    if (ov > v || (ov == v && oi < i)) { v = ov; i = oi; }
    const int gm = gmBase + tid;
    idxbuf[gm] = i;
    idx_out[gm] = (float)i;
  }
}

// ---------------------------------------------------------------------------
// Kernel 3a: zero the loss slot (d_out is poisoned by the harness)
// ---------------------------------------------------------------------------
__global__ void vq_zero_loss_kernel(float* loss_out) { *loss_out = 0.0f; }

// ---------------------------------------------------------------------------
// Kernel 3b: gather z_q = emb[idx] into [B,D,T] + fused MSE loss
// ---------------------------------------------------------------------------
__global__ __launch_bounds__(256) void vq_gather_loss_kernel(const float* __restrict__ z_e,
                                                             const float* __restrict__ emb,
                                                             const int* __restrict__ idxbuf,
                                                             float* __restrict__ zq_out,
                                                             float* __restrict__ loss_out) {
  __shared__ int sidx[64];
  __shared__ float sred[256];
  const int tid = threadIdx.x;
  const int b  = blockIdx.x >> 4;
  const int t0 = (blockIdx.x & 15) << 6;
  if (tid < 64) sidx[tid] = idxbuf[b * T_DIM + t0 + tid];
  __syncthreads();
  const int tt = tid & 63;
  const int d0 = tid >> 6;
  const int t = t0 + tt;
  float acc = 0.0f;
  for (int d = d0; d < D_DIM; d += 4) {
    const float q = emb[(size_t)sidx[tt] * D_DIM + d];
    const size_t off = (size_t)b * (D_DIM * T_DIM) + (size_t)d * T_DIM + t;
    const float x = z_e[off];
    zq_out[off] = q;  // straight-through forward value == z_q
    const float diff = q - x;
    acc += diff * diff;
  }
  sred[tid] = acc;
  __syncthreads();
  for (int s = 128; s > 0; s >>= 1) {
    if (tid < s) sred[tid] += sred[tid + s];
    __syncthreads();
  }
  if (tid == 0) atomicAdd(loss_out, sred[0] * (1.0f / (float)BDT));
}

// ---------------------------------------------------------------------------
extern "C" void kernel_launch(void* const* d_in, const int* in_sizes, int n_in,
                              void* d_out, int out_size, void* d_ws, size_t ws_size,
                              hipStream_t stream) {
  const float* z_e = (const float*)d_in[0];  // [B, D, T] f32
  const float* emb = (const float*)d_in[1];  // [K, D]   f32

  float* out      = (float*)d_out;
  float* zq_out   = out;
  float* loss_out = out + BDT;
  float* idx_out  = loss_out + 1;

  char* ws = (char*)d_ws;
  _Float16* xT     = (_Float16*)(ws);                                     // 16 MB
  _Float16* ef16   = (_Float16*)(ws + (size_t)16 * 1024 * 1024);          // 4 MB
  float*    cbuf   = (float*)(ws + (size_t)20 * 1024 * 1024);             // 32 KB
  int*      idxbuf = (int*)(ws + (size_t)20 * 1024 * 1024 + 64 * 1024);   // 128 KB

  vq_xpose_kernel<<<dim3(T_DIM / 32, D_DIM / 32, B_DIM), dim3(32, 8), 0, stream>>>(z_e, xT);
  vq_embprep_kernel<<<K_CODES, 256, 0, stream>>>(emb, ef16, cbuf);
  vq_argmax_kernel<<<BT / BLOCK_M, 256, 0, stream>>>(xT, ef16, cbuf, idxbuf, idx_out);
  vq_zero_loss_kernel<<<1, 1, 0, stream>>>(loss_out);
  vq_gather_loss_kernel<<<B_DIM * (T_DIM / 64), 256, 0, stream>>>(z_e, emb, idxbuf, zq_out,
                                                                  loss_out);
}
